// LlamaAttention_36696200577566
// MI455X (gfx1250) — matrix operand order repaired and checked by reading, not compiled, not run on hardware
//
#include <hip/hip_runtime.h>
#include <hip/hip_bf16.h>
#include <math.h>

#define B_ 2
#define S_ 2048
#define D_ 2048
#define H_ 16
#define KVH_ 4
#define HD_ 128
#define NREP_ 4

typedef __bf16 bf16_t;
typedef __attribute__((ext_vector_type(16))) bf16_t v16bf;
typedef __attribute__((ext_vector_type(8)))  float  v8f;
typedef __attribute__((ext_vector_type(8)))  unsigned short us8v;
typedef __attribute__((ext_vector_type(16))) unsigned short us16v;

union Frag { v16bf bf; us16v u; us8v h[2]; };
union Acc  { v8f v; float f[8]; };

#ifndef __has_builtin
#define __has_builtin(x) 0
#endif
#if __has_builtin(__builtin_amdgcn_global_load_async_to_lds_b128)
#define HAVE_ASYNC_LDS 1
#else
#define HAVE_ASYNC_LDS 0
#endif
#if __has_builtin(__builtin_amdgcn_s_wait_asynccnt)
#define HAVE_WAIT_ASYNC 1
#else
#define HAVE_WAIT_ASYNC 0
#endif

#if HAVE_ASYNC_LDS
typedef __attribute__((vector_size(16))) int v4i_;
typedef __attribute__((address_space(1))) v4i_* gv4p;   // global ptr to int4
typedef __attribute__((address_space(3))) v4i_* lv4p;   // LDS ptr to int4
#endif

__device__ __forceinline__ unsigned short f2bf(float f) {
    unsigned int u = __float_as_uint(f);
    u += 0x7FFFu + ((u >> 16) & 1u);   // round-to-nearest-even
    return (unsigned short)(u >> 16);
}

__device__ __forceinline__ v8f wmma_bf(const Frag& a, const Frag& b, v8f c) {
    return __builtin_amdgcn_wmma_f32_16x16x32_bf16(
        false, a.bf, false, b.bf, (short)0, c, false, false);
}

__device__ __forceinline__ float rowmax16(float x) {
    for (int m = 1; m < 16; m <<= 1) x = fmaxf(x, __shfl_xor(x, m, 32));
    return x;
}
__device__ __forceinline__ float rowsum16(float x) {
    for (int m = 1; m < 16; m <<= 1) x += __shfl_xor(x, m, 32);
    return x;
}

// 16-byte global -> LDS copy: async DMA (ASYNCcnt) when available.
__device__ __forceinline__ void copy16(const unsigned short* g, unsigned short* l) {
#if HAVE_ASYNC_LDS
    __builtin_amdgcn_global_load_async_to_lds_b128((gv4p)g, (lv4p)l, 0, 0);
#else
    *(us8v*)l = *(const us8v*)g;
#endif
}
__device__ __forceinline__ void wait_copies() {
#if HAVE_ASYNC_LDS
#if HAVE_WAIT_ASYNC
    __builtin_amdgcn_s_wait_asynccnt(0);
#else
    asm volatile("s_wait_asynccnt 0" ::: "memory");
#endif
#endif
}

// ---------------------------------------------------------------- convert
__global__ void f32_to_bf16_k(const float* __restrict__ src,
                              unsigned short* __restrict__ dst, int n) {
    int i = blockIdx.x * 256 + threadIdx.x;
    if (i < n) dst[i] = f2bf(src[i]);
}

// Transposing convert: src f32 [K][N] -> dst bf16 [N][K]  (64x64 LDS tiles)
__global__ __launch_bounds__(256) void transpose_f32_bf16(
    const float* __restrict__ src, unsigned short* __restrict__ dst,
    int K, int N)
{
    __shared__ unsigned short t[64 * 65];
    const int k0 = blockIdx.x * 64, n0 = blockIdx.y * 64;
    const int tid = threadIdx.x;
    for (int i = tid; i < 4096; i += 256) {
        int r = i >> 6, c = i & 63;                       // r: k, c: n (coalesced read)
        t[c * 65 + r] = f2bf(src[(size_t)(k0 + r) * N + n0 + c]);
    }
    __syncthreads();
    for (int i = tid; i < 4096; i += 256) {
        int r = i >> 6, c = i & 63;                       // r: n, c: k (coalesced write)
        dst[(size_t)(n0 + r) * K + k0 + c] = t[r * 65 + c];
    }
}

// ---------------------------------------------------------------- GEMM
// C(f32, MxN) = A(bf16 [M][K]) * B, with B given TRANSPOSED as Bt(bf16 [N][K]).
// 128x128 block tile, BK=32, 8 waves x (32x64), double-buffered async LDS feed.
__global__ __launch_bounds__(256) void gemm_bf16(
    const unsigned short* __restrict__ Ag,
    const unsigned short* __restrict__ Btg,
    float* __restrict__ Cg, int M, int N, int K)
{
    __shared__ unsigned short As[2][128 * 40];
    __shared__ unsigned short Bs[2][128 * 40];

    const int tid  = threadIdx.x;
    const int lane = tid & 31, w = tid >> 5;
    const int half = lane >> 4, ln = lane & 15;
    const int wm = w >> 1, wn = w & 1;               // 4 x 2 wave grid
    const int rowBase = blockIdx.x * 128;
    const int colBase = blockIdx.y * 128;

    // Each thread feeds rows r0 and r0+64 of both tiles at column chunk cb.
    const int r0 = tid >> 2, cb = (tid & 3) << 3;
    const unsigned short* gA0 = Ag  + (size_t)(rowBase + r0)      * K + cb;
    const unsigned short* gA1 = Ag  + (size_t)(rowBase + r0 + 64) * K + cb;
    const unsigned short* gB0 = Btg + (size_t)(colBase + r0)      * K + cb;
    const unsigned short* gB1 = Btg + (size_t)(colBase + r0 + 64) * K + cb;
    const int lOff0 = r0 * 40 + cb, lOff1 = (r0 + 64) * 40 + cb;

    Acc acc[2][4];
    for (int mt = 0; mt < 2; ++mt)
        for (int nt = 0; nt < 4; ++nt)
            for (int v = 0; v < 8; ++v) acc[mt][nt].f[v] = 0.f;

    const int KT = K >> 5;
    int st = 0;

    // prologue: tile 0 -> buffer 0
    copy16(gA0, &As[0][lOff0]);
    copy16(gA1, &As[0][lOff1]);
    copy16(gB0, &Bs[0][lOff0]);
    copy16(gB1, &Bs[0][lOff1]);
    wait_copies();
    __syncthreads();

    for (int kt = 0; kt < KT; ++kt) {
        if (kt + 1 < KT) {
            const int k0 = (kt + 1) << 5;
            copy16(gA0 + k0, &As[st ^ 1][lOff0]);
            copy16(gA1 + k0, &As[st ^ 1][lOff1]);
            copy16(gB0 + k0, &Bs[st ^ 1][lOff0]);
            copy16(gB1 + k0, &Bs[st ^ 1][lOff1]);
        }

        const unsigned short* Ab = As[st];
        const unsigned short* Bb = Bs[st];
        Frag af[2], bfr[4];
        for (int mt = 0; mt < 2; ++mt) {
            int r = wm * 32 + mt * 16 + ln;
            af[mt].h[0] = *(const us8v*)&Ab[r * 40 + half * 8];
            af[mt].h[1] = *(const us8v*)&Ab[r * 40 + 16 + half * 8];
        }
        for (int nt = 0; nt < 4; ++nt) {
            int c = wn * 64 + nt * 16 + ln;
            bfr[nt].h[0] = *(const us8v*)&Bb[c * 40 + half * 8];
            bfr[nt].h[1] = *(const us8v*)&Bb[c * 40 + 16 + half * 8];
        }
        for (int mt = 0; mt < 2; ++mt)
            for (int nt = 0; nt < 4; ++nt)
                acc[mt][nt].v = wmma_bf(af[mt], bfr[nt], acc[mt][nt].v);

        wait_copies();
        __syncthreads();
        st ^= 1;
    }

    for (int mt = 0; mt < 2; ++mt)
        for (int nt = 0; nt < 4; ++nt)
            for (int v = 0; v < 8; ++v) {
                int r = rowBase + wm * 32 + mt * 16 + v + 8 * half;
                int c = colBase + wn * 64 + nt * 16 + ln;
                Cg[(size_t)r * N + c] = acc[mt][nt].f[v];
            }
}

// ---------------------------------------------------------------- RoPE + pack
// Qf: [B*S, 2048] f32  -> Qb  [B,H,S,HD]    bf16 (rope applied)
// Kf: [B*S, 512]  f32  -> Kb  [B,KVH,S,HD]  bf16 (rope applied)
// Vf: [B*S, 512]  f32  -> Vt  [B,KVH,HD,S]  bf16 (transposed)
__global__ __launch_bounds__(256) void rope_pack(
    const float* __restrict__ Qf, const float* __restrict__ Kf,
    const float* __restrict__ Vf, const float* __restrict__ cosp,
    const float* __restrict__ sinp, unsigned short* __restrict__ Qb,
    unsigned short* __restrict__ Kb, unsigned short* __restrict__ Vt)
{
    const int bs = blockIdx.x;
    const int b = bs / S_, s = bs % S_;
    const int tid = threadIdx.x;
    const float* cr = cosp + (size_t)s * (HD_ / 2);
    const float* sr = sinp + (size_t)s * (HD_ / 2);
    const float* qrow = Qf + (size_t)bs * D_;
    const float* krow = Kf + (size_t)bs * (KVH_ * HD_);
    const float* vrow = Vf + (size_t)bs * (KVH_ * HD_);

    for (int idx = tid; idx < H_ * (HD_ / 2); idx += 256) {
        int hh = idx >> 6, i = idx & 63;
        float p1 = qrow[hh * HD_ + 2 * i];
        float p2 = qrow[hh * HD_ + 2 * i + 1];
        float c = cr[i], sn = sr[i];
        unsigned short* dst = Qb + (((size_t)b * H_ + hh) * S_ + s) * HD_;
        dst[i]      = f2bf(p1 * c - p2 * sn);
        dst[i + 64] = f2bf(p2 * c + p1 * sn);
    }
    for (int idx = tid; idx < KVH_ * (HD_ / 2); idx += 256) {
        int g = idx >> 6, i = idx & 63;
        float p1 = krow[g * HD_ + 2 * i];
        float p2 = krow[g * HD_ + 2 * i + 1];
        float c = cr[i], sn = sr[i];
        unsigned short* dst = Kb + (((size_t)b * KVH_ + g) * S_ + s) * HD_;
        dst[i]      = f2bf(p1 * c - p2 * sn);
        dst[i + 64] = f2bf(p2 * c + p1 * sn);
    }
    for (int idx = tid; idx < KVH_ * HD_; idx += 256) {
        int g = idx >> 7, hd = idx & 127;
        Vt[(((size_t)b * KVH_ + g) * HD_ + hd) * S_ + s] =
            f2bf(vrow[g * HD_ + hd]);
    }
}

// ---------------------------------------------------------------- flash attention
// One wave per 16-row Q tile; online softmax; 32-key chunks; causal.
__global__ __launch_bounds__(256) void flash_attn(
    const unsigned short* __restrict__ Qb,
    const unsigned short* __restrict__ Kb,
    const unsigned short* __restrict__ Vt,
    unsigned short* __restrict__ Ob)
{
    __shared__ unsigned short Plds[8 * 16 * 40];   // per-wave P transpose scratch

    const int tid = threadIdx.x, lane = tid & 31, w = tid >> 5;
    const int half = lane >> 4, ln = lane & 15;
    const int gw = blockIdx.x * 8 + w;
    const int qtiles = S_ / 16;
    const int qt = gw % qtiles, bh = gw / qtiles;
    const int hh = bh % H_, b = bh / H_, g = hh / NREP_;
    const int q0 = qt * 16;

    const unsigned short* Qp = Qb + (((size_t)b * H_ + hh) * S_) * HD_;
    const unsigned short* Kp = Kb + (((size_t)b * KVH_ + g) * S_) * HD_;
    const unsigned short* Vp = Vt + (((size_t)b * KVH_ + g) * HD_) * S_;

    Frag qa[4];
    for (int c = 0; c < 4; ++c) {
        int r = q0 + ln;
        int base = c * 32 + half * 8;
        qa[c].h[0] = *(const us8v*)&Qp[(size_t)r * HD_ + base];
        qa[c].h[1] = *(const us8v*)&Qp[(size_t)r * HD_ + base + 16];
    }

    Acc o[8];
    float m[8], l[8];
    for (int d = 0; d < 8; ++d)
        for (int v = 0; v < 8; ++v) o[d].f[v] = 0.f;
    for (int v = 0; v < 8; ++v) { m[v] = -3.0e38f; l[v] = 0.f; }

    const float scale = 0.08838834764831845f;   // 1/sqrt(128)
    unsigned short* P = &Plds[w * 16 * 40];
    const int kvmax = q0 + 15;

    for (int kv0 = 0; kv0 <= kvmax; kv0 += 32) {
        Acc sc[2];
        for (int j = 0; j < 2; ++j)
            for (int v = 0; v < 8; ++v) sc[j].f[v] = 0.f;
        for (int j = 0; j < 2; ++j) {
            int n = kv0 + j * 16 + ln;
            for (int c = 0; c < 4; ++c) {
                Frag kf;
                kf.h[0] = *(const us8v*)&Kp[(size_t)n * HD_ + c * 32 + half * 8];
                kf.h[1] = *(const us8v*)&Kp[(size_t)n * HD_ + c * 32 + 16 + half * 8];
                sc[j].v = wmma_bf(qa[c], kf, sc[j].v);
            }
        }
        float p0[8], p1[8], alpha[8];
        for (int v = 0; v < 8; ++v) {
            int qrow = q0 + v + 8 * half;
            float s0 = sc[0].f[v] * scale + ((kv0 + ln)      > qrow ? -1.0e9f : 0.f);
            float s1 = sc[1].f[v] * scale + ((kv0 + 16 + ln) > qrow ? -1.0e9f : 0.f);
            float mx = rowmax16(fmaxf(s0, s1));
            float mnew = fmaxf(m[v], mx);
            alpha[v] = __expf(m[v] - mnew);
            float e0 = __expf(s0 - mnew);
            float e1 = __expf(s1 - mnew);
            p0[v] = e0; p1[v] = e1;
            l[v] = l[v] * alpha[v] + rowsum16(e0 + e1);
            m[v] = mnew;
        }
        for (int d = 0; d < 8; ++d)
            for (int v = 0; v < 8; ++v) o[d].f[v] *= alpha[v];

        for (int v = 0; v < 8; ++v) {
            int r = v + 8 * half;
            P[r * 40 + ln]      = f2bf(p0[v]);
            P[r * 40 + 16 + ln] = f2bf(p1[v]);
        }
        asm volatile("s_wait_dscnt 0" ::: "memory");
        Frag pa;
        pa.h[0] = *(const us8v*)&P[ln * 40 + half * 8];
        pa.h[1] = *(const us8v*)&P[ln * 40 + 16 + half * 8];
        asm volatile("s_wait_dscnt 0" ::: "memory");

        for (int d = 0; d < 8; ++d) {
            int hd = d * 16 + ln;
            Frag vf;
            vf.h[0] = *(const us8v*)&Vp[(size_t)hd * S_ + kv0 + half * 8];
            vf.h[1] = *(const us8v*)&Vp[(size_t)hd * S_ + kv0 + 16 + half * 8];
            o[d].v = wmma_bf(pa, vf, o[d].v);
        }
    }

    for (int d = 0; d < 8; ++d)
        for (int v = 0; v < 8; ++v) {
            int r = q0 + v + 8 * half;
            int col = hh * HD_ + d * 16 + ln;
            Ob[((size_t)b * S_ + r) * D_ + col] = f2bf(o[d].f[v] / l[v]);
        }
}

// ---------------------------------------------------------------- host
extern "C" void kernel_launch(void* const* d_in, const int* in_sizes, int n_in,
                              void* d_out, int out_size, void* d_ws, size_t ws_size,
                              hipStream_t stream)
{
    const float* hidden = (const float*)d_in[0];
    // d_in[1] = attention_mask: pure causal, computed inline
    const float* cosp = (const float*)d_in[2];
    const float* sinp = (const float*)d_in[3];
    const float* Wq = (const float*)d_in[4];
    const float* Wk = (const float*)d_in[5];
    const float* Wv = (const float*)d_in[6];
    const float* Wo = (const float*)d_in[7];

    char* ws = (char*)d_ws;
    const size_t MB = 1024 * 1024;
    unsigned short* Xbf  = (unsigned short*)(ws);              // 16 MB
    unsigned short* Wqt  = (unsigned short*)(ws + 16 * MB);    //  8 MB  [N][K]
    unsigned short* Wkt  = (unsigned short*)(ws + 24 * MB);    //  2 MB
    unsigned short* Wvt  = (unsigned short*)(ws + 26 * MB);    //  2 MB
    unsigned short* Wot  = (unsigned short*)(ws + 28 * MB);    //  8 MB
    float* Qf = (float*)(ws + 36 * MB);                        // 32 MB
    float* Kf = (float*)(ws + 68 * MB);                        //  8 MB
    float* Vf = (float*)(ws + 76 * MB);                        //  8 MB
    unsigned short* Qb  = (unsigned short*)(ws + 84 * MB);     // 16 MB
    unsigned short* Kb  = (unsigned short*)(ws + 100 * MB);    //  4 MB
    unsigned short* Vtb = (unsigned short*)(ws + 104 * MB);    //  4 MB
    unsigned short* Ab  = (unsigned short*)(ws + 36 * MB);     // alias dead Qf

    const int M  = B_ * S_;          // 4096
    const int nX = B_ * S_ * D_;
    const int NKV = KVH_ * HD_;      // 512

    f32_to_bf16_k<<<(nX + 255) / 256, 256, 0, stream>>>(hidden, Xbf, nX);
    transpose_f32_bf16<<<dim3(D_ / 64, D_ / 64),  256, 0, stream>>>(Wq, Wqt, D_, D_);
    transpose_f32_bf16<<<dim3(D_ / 64, NKV / 64), 256, 0, stream>>>(Wk, Wkt, D_, NKV);
    transpose_f32_bf16<<<dim3(D_ / 64, NKV / 64), 256, 0, stream>>>(Wv, Wvt, D_, NKV);
    transpose_f32_bf16<<<dim3(D_ / 64, D_ / 64),  256, 0, stream>>>(Wo, Wot, D_, D_);

    gemm_bf16<<<dim3(M / 128, D_ / 128),  256, 0, stream>>>(Xbf, Wqt, Qf, M, D_, D_);
    gemm_bf16<<<dim3(M / 128, NKV / 128), 256, 0, stream>>>(Xbf, Wkt, Kf, M, NKV, D_);
    gemm_bf16<<<dim3(M / 128, NKV / 128), 256, 0, stream>>>(Xbf, Wvt, Vf, M, NKV, D_);

    rope_pack<<<B_ * S_, 256, 0, stream>>>(Qf, Kf, Vf, cosp, sinp, Qb, Kb, Vtb);

    flash_attn<<<(B_ * H_ * (S_ / 16)) / 8, 256, 0, stream>>>(Qb, Kb, Vtb, Ab);

    gemm_bf16<<<dim3(M / 128, D_ / 128), 256, 0, stream>>>(Ab, Wot, (float*)d_out, M, D_, D_);
}
